// KalmanNetNN_78975858639059
// MI455X (gfx1250) — matrix operand order repaired and checked by reading, not compiled, not run on hardware
//
// KalmanNet RNN for MI455X (gfx1250): bf16 weights resident in 192MB L2,
// WMMA-based GEMVs with split-K load balancing, persistent kernel with
// grid-wide barriers (5 per step). B operands are broadcast across lanes
// (column 0 of D only depends on column 0 of B), so the inner loop has no
// exec-mask divergence.
#include <hip/hip_runtime.h>
#include <hip/hip_bf16.h>

typedef __attribute__((ext_vector_type(16))) __bf16 v16bf;
typedef __attribute__((ext_vector_type(8)))  float  v8f;

namespace kn {
constexpr int MM = 4, NN = 48, TT = 512, DIN = 52;
constexpr int H1 = 4160;            // layer-1 width
constexpr int HID = 2320;           // GRU hidden
constexpr int HIDP = 2336;          // HID padded to %32
constexpr int G3 = 3 * HID;         // 6960 gate rows
constexpr int H2 = 768;
constexpr int DOUT = 192;           // M*N
constexpr int K1P = 64;             // DIN padded to %32
constexpr int NBLK = 128, NTHR = 256;
constexpr int L1T = H1 / 16;        // 260 tiles
constexpr int GHT = G3 / 16;        // 435 tiles
constexpr int GIT = G3 / 16;        // 435 tiles
constexpr int L2T = H2 / 16;        // 48 tiles
constexpr int KGT = DOUT / 16;      // 12 tiles
constexpr int S_GH = 2, S_GI = 2, S_L2 = 8, S_KG = 8;  // split-K factors
}

union ABits { v16bf v; uint4 u[2]; };

__device__ __forceinline__ v16bf ld_a(const __bf16* p) {
  ABits a;
  a.u[0] = *(const uint4*)(p);
  a.u[1] = *(const uint4*)(p + 8);
  return a.v;
}

// ---------------------------------------------------------------------------
// Grid-wide barrier (sense-reversing, agent-scope atomics, s_sleep spin)
// ---------------------------------------------------------------------------
__device__ __forceinline__ void grid_sync(unsigned* cnt, unsigned* gen) {
  __threadfence();
  __syncthreads();
  if (threadIdx.x == 0) {
    unsigned g = __hip_atomic_load(gen, __ATOMIC_RELAXED, __HIP_MEMORY_SCOPE_AGENT);
    unsigned prev = __hip_atomic_fetch_add(cnt, 1u, __ATOMIC_ACQ_REL, __HIP_MEMORY_SCOPE_AGENT);
    if (prev == gridDim.x - 1u) {
      __hip_atomic_store(cnt, 0u, __ATOMIC_RELAXED, __HIP_MEMORY_SCOPE_AGENT);
      __hip_atomic_store(gen, g + 1u, __ATOMIC_RELEASE, __HIP_MEMORY_SCOPE_AGENT);
    } else {
      while (__hip_atomic_load(gen, __ATOMIC_ACQUIRE, __HIP_MEMORY_SCOPE_AGENT) == g) {
        __builtin_amdgcn_s_sleep(2);
      }
    }
  }
  __syncthreads();
  __threadfence();
}

// ---------------------------------------------------------------------------
// Split-K WMMA GEMV. One wave per (16-row tile, K-slice) unit.
// B is broadcast (every column = the vector slice), so all D columns hold the
// same dot products; rows are extracted from lanes 0 and 16.
// OUTMODE 0: raw fp32 partial (no bias/act) -> outF[s*rowsTot + row]
// OUTMODE 1: bf16 relu(acc + bias)          -> outB[row]       (S must be 1)
// ---------------------------------------------------------------------------
template<int OUTMODE>
__device__ __forceinline__ void wmma_gemv_sk(
    const __bf16* __restrict__ W, int ldw,
    const __bf16* __restrict__ xv, const float* __restrict__ bias,
    float* __restrict__ outF, __bf16* __restrict__ outB,
    int ntiles, int rowsTot, int K, int S,
    int gw, int gws, int lane)
{
  const int m  = lane & 15;
  const int hi = lane >> 4;
  const int nch = K >> 5;
  const int nunits = ntiles * S;
  for (int u = gw; u < nunits; u += gws) {
    const int tile = u / S, s = u - tile * S;
    const int c0 = (nch * s) / S, c1 = (nch * (s + 1)) / S;
    const int kbeg = c0 << 5, kend = c1 << 5;
    const __bf16* wrow = W + (size_t)(tile * 16 + m) * (size_t)ldw + hi * 16;
    const __bf16* xb   = xv + hi * 16;     // per-half-wave K slice (broadcast)
    v8f acc0 = {}, acc1 = {}, acc2 = {}, acc3 = {};
    int k0 = kbeg;
    for (; k0 + 128 <= kend; k0 += 128) {      // 4 independent WMMA chains
      __builtin_prefetch(wrow + k0 + 1024, 0, 3);   // WGP-scope prefetch
      v16bf a0 = ld_a(wrow + k0);
      v16bf a1 = ld_a(wrow + k0 + 32);
      v16bf a2 = ld_a(wrow + k0 + 64);
      v16bf a3 = ld_a(wrow + k0 + 96);
      v16bf b0 = *(const v16bf*)(xb + k0);
      v16bf b1 = *(const v16bf*)(xb + k0 + 32);
      v16bf b2 = *(const v16bf*)(xb + k0 + 64);
      v16bf b3 = *(const v16bf*)(xb + k0 + 96);
      acc0 = __builtin_amdgcn_wmma_f32_16x16x32_bf16(false, a0, false, b0, (short)0, acc0, false, false);
      acc1 = __builtin_amdgcn_wmma_f32_16x16x32_bf16(false, a1, false, b1, (short)0, acc1, false, false);
      acc2 = __builtin_amdgcn_wmma_f32_16x16x32_bf16(false, a2, false, b2, (short)0, acc2, false, false);
      acc3 = __builtin_amdgcn_wmma_f32_16x16x32_bf16(false, a3, false, b3, (short)0, acc3, false, false);
    }
    for (; k0 < kend; k0 += 32) {              // K-chunk remainder
      v16bf a0 = ld_a(wrow + k0);
      v16bf b0 = *(const v16bf*)(xb + k0);
      acc0 = __builtin_amdgcn_wmma_f32_16x16x32_bf16(false, a0, false, b0, (short)0, acc0, false, false);
    }
    acc0 = (acc0 + acc1) + (acc2 + acc3);
    if (m == 0) {  // rows live in lanes 0 (rows 0..7) and 16 (rows 8..15)
      const int rbase = tile * 16 + hi * 8;
      #pragma unroll
      for (int r = 0; r < 8; ++r) {
        if (OUTMODE == 0) {
          outF[(size_t)s * rowsTot + rbase + r] = acc0[r];
        } else {
          float v = fmaxf(acc0[r] + bias[rbase + r], 0.f);
          outB[rbase + r] = (__bf16)v;
        }
      }
    }
  }
}

// ---------------------------------------------------------------------------
// Phase V: kg partials = W3 @ relu(sum_s l2p[s] + b2), split-K.
// All lanes build the broadcast B (same addresses per half-wave -> cache
// broadcast); fuses the l2 partial reduction + bias + ReLU.
// ---------------------------------------------------------------------------
__device__ __forceinline__ void wmma_gemv_kg(
    const __bf16* __restrict__ W3, const float* __restrict__ l2p,
    const float* __restrict__ b2v, float* __restrict__ kgp,
    int gw, int gws, int lane)
{
  using namespace kn;
  const int m  = lane & 15;
  const int hi = lane >> 4;
  const int nch = H2 >> 5;             // 24
  const int nunits = KGT * S_KG;       // 96
  for (int u = gw; u < nunits; u += gws) {
    const int tile = u / S_KG, s = u - tile * S_KG;
    const int c0 = (nch * s) / S_KG, c1 = (nch * (s + 1)) / S_KG;
    const __bf16* wrow = W3 + (size_t)(tile * 16 + m) * H2 + hi * 16;
    v8f acc = {};
    for (int c = c0; c < c1; ++c) {
      const int k0 = c << 5;
      v16bf a = ld_a(wrow + k0);
      v16bf b;
      #pragma unroll
      for (int e = 0; e < 16; ++e) {
        const int k = k0 + hi * 16 + e;
        float v = b2v[k];
        #pragma unroll
        for (int p = 0; p < S_L2; ++p) v += l2p[p * H2 + k];
        b[e] = (__bf16)fmaxf(v, 0.f);
      }
      acc = __builtin_amdgcn_wmma_f32_16x16x32_bf16(false, a, false, b, (short)0, acc, false, false);
    }
    if (m == 0) {
      const int rbase = tile * 16 + hi * 8;
      #pragma unroll
      for (int r = 0; r < 8; ++r) kgp[s * DOUT + rbase + r] = acc[r];
    }
  }
}

// ---------------------------------------------------------------------------
// Init: fp32 -> bf16 weight conversion with K padding, h state init, barriers
// ---------------------------------------------------------------------------
__global__ void knet_init(
    const float* __restrict__ W1, const float* __restrict__ Wih,
    const float* __restrict__ Whh, const float* __restrict__ W2,
    const float* __restrict__ W3, const float* __restrict__ h0,
    __bf16* __restrict__ W1p, __bf16* __restrict__ Wihb,
    __bf16* __restrict__ Whhp, __bf16* __restrict__ W2p,
    __bf16* __restrict__ W3b, float* __restrict__ hf,
    __bf16* __restrict__ hbf, unsigned* __restrict__ cnt,
    unsigned* __restrict__ gen)
{
  using namespace kn;
  size_t gid = (size_t)blockIdx.x * blockDim.x + threadIdx.x;
  size_t gs  = (size_t)gridDim.x * blockDim.x;
  if (gid == 0) { *cnt = 0u; *gen = 0u; }
  for (size_t i = gid; i < (size_t)H1 * K1P; i += gs) {
    int r = (int)(i >> 6), k = (int)(i & 63);
    W1p[i] = (__bf16)(k < DIN ? W1[(size_t)r * DIN + k] : 0.f);
  }
  for (size_t i = gid; i < (size_t)G3 * H1; i += gs)
    Wihb[i] = (__bf16)Wih[i];
  for (size_t i = gid; i < (size_t)G3 * HIDP; i += gs) {
    int r = (int)(i / HIDP), k = (int)(i % HIDP);
    Whhp[i] = (__bf16)(k < HID ? Whh[(size_t)r * HID + k] : 0.f);
  }
  for (size_t i = gid; i < (size_t)H2 * HIDP; i += gs) {
    int r = (int)(i / HIDP), k = (int)(i % HIDP);
    W2p[i] = (__bf16)(k < HID ? W2[(size_t)r * HID + k] : 0.f);
  }
  for (size_t i = gid; i < (size_t)DOUT * H2; i += gs)
    W3b[i] = (__bf16)W3[i];
  for (size_t i = gid; i < (size_t)HIDP; i += gs) {
    float v = (i < (size_t)HID) ? h0[i] : 0.f;
    hf[i]  = v;
    hbf[i] = (__bf16)v;
  }
}

// ---------------------------------------------------------------------------
// Persistent kernel: all T=512 steps, 5 grid barriers per step
// ---------------------------------------------------------------------------
__global__ __launch_bounds__(kn::NTHR, 1) void knet_persist(
    const float* __restrict__ Amat, const float* __restrict__ Cmat,
    const float* __restrict__ x0,   const float* __restrict__ y,
    const float* __restrict__ b1,   const float* __restrict__ bih,
    const float* __restrict__ bhh,  const float* __restrict__ b2,
    const float* __restrict__ b3,
    const __bf16* __restrict__ W1p, const __bf16* __restrict__ Wihb,
    const __bf16* __restrict__ Whhp, const __bf16* __restrict__ W2p,
    const __bf16* __restrict__ W3b,
    __bf16* __restrict__ l1b,
    float* __restrict__ gip, float* __restrict__ ghp,
    float* __restrict__ l2p, float* __restrict__ kgp,
    float* __restrict__ hf, __bf16* __restrict__ hbf,
    unsigned* __restrict__ cnt, unsigned* __restrict__ gen,
    float* __restrict__ out)
{
  using namespace kn;
  __shared__ float sPrior[MM], sPost[MM], sDx[MM], sTmp[MM], sDy[NN], sNrm[2];
  __shared__ alignas(64) __bf16 sKnet[K1P];

  const int tid  = threadIdx.x;
  const int lane = tid & 31;
  const int gw   = blockIdx.x * (NTHR / 32) + (tid >> 5);
  const int GW   = NBLK * (NTHR / 32);

  if (tid < MM) { sPrior[tid] = x0[tid]; sPost[tid] = x0[tid]; }
  __syncthreads();

  for (int t = 0; t < TT; ++t) {
    // ---- scalar Kalman prep, redundant per block (LDS-resident state) ----
    if (t > 0) {
      if (tid < MM) {  // posterior t-1: x_post = x_prior + kg @ dy (sum kg partials)
        float v = sPrior[tid];
        for (int n = 0; n < NN; ++n) {
          float kgv = b3[tid * NN + n];
          #pragma unroll
          for (int s = 0; s < S_KG; ++s) kgv += kgp[s * DOUT + tid * NN + n];
          v += kgv * 1e-4f * sDy[n];
        }
        sTmp[tid] = v;
      }
      __syncthreads();
      if (tid < MM) {
        if (blockIdx.x == 0) out[tid * TT + (t - 1)] = sTmp[tid];
        sPost[tid] = sTmp[tid];
      }
      __syncthreads();
    }
    if (tid < MM) {  // new prior + dx (dx uses the old prior)
      float v = 0.f;
      #pragma unroll
      for (int j = 0; j < MM; ++j) v += Amat[tid * MM + j] * sPost[j];
      sTmp[tid] = v;
      sDx[tid]  = sPost[tid] - sPrior[tid];
    }
    __syncthreads();
    if (tid < MM) sPrior[tid] = sTmp[tid];
    __syncthreads();
    if (tid < NN) {  // innovation: dy = y_t - C @ [x_prior; 1]
      float v = Cmat[tid * (MM + 1) + MM];
      #pragma unroll
      for (int j = 0; j < MM; ++j) v += Cmat[tid * (MM + 1) + j] * sPrior[j];
      sDy[tid] = y[tid * TT + t] - v;
    }
    __syncthreads();
    if (tid == 0) {  // L2 norms (48- and 4-dim: cheap serial)
      float s = 0.f;
      for (int n = 0; n < NN; ++n) s += sDy[n] * sDy[n];
      sNrm[0] = 1.f / fmaxf(sqrtf(s), 1e-12f);
      s = 0.f;
      for (int j = 0; j < MM; ++j) s += sDx[j] * sDx[j];
      sNrm[1] = 1.f / fmaxf(sqrtf(s), 1e-12f);
    }
    __syncthreads();
    if (tid < K1P) {  // KGainNet input, bf16, zero-padded to 64
      float v = (tid < NN)      ? sDy[tid] * sNrm[0]
              : (tid < NN + MM) ? sDx[tid - NN] * sNrm[1] : 0.f;
      sKnet[tid] = (__bf16)v;
    }
    __syncthreads();

    // ---- Phase I: l1 = relu(W1 @ knet + b1); gh partials = W_hh @ h_prev --
    wmma_gemv_sk<1>(W1p,  K1P,  sKnet, b1, nullptr, l1b, L1T, H1, K1P, 1, gw, GW, lane);
    wmma_gemv_sk<0>(Whhp, HIDP, hbf, nullptr, ghp, nullptr, GHT, G3, HIDP, S_GH, gw, GW, lane);
    grid_sync(cnt, gen);

    // ---- Phase II: gi partials = W_ih @ l1 (58 MB bf16 from L2) ----
    wmma_gemv_sk<0>(Wihb, H1, l1b, nullptr, gip, nullptr, GIT, G3, H1, S_GI, gw, GW, lane);
    grid_sync(cnt, gen);

    // ---- Phase III: GRU combine (sums split-K partials + biases) ----
    for (int j = blockIdx.x * NTHR + tid; j < HID; j += NBLK * NTHR) {
      float ir = gip[j]           + gip[G3 + j]           + bih[j];
      float iz = gip[HID + j]     + gip[G3 + HID + j]     + bih[HID + j];
      float in_= gip[2 * HID + j] + gip[G3 + 2 * HID + j] + bih[2 * HID + j];
      float hr = ghp[j]           + ghp[G3 + j]           + bhh[j];
      float hz = ghp[HID + j]     + ghp[G3 + HID + j]     + bhh[HID + j];
      float hn = ghp[2 * HID + j] + ghp[G3 + 2 * HID + j] + bhh[2 * HID + j];
      float r  = 1.f / (1.f + __expf(-(ir + hr)));
      float z  = 1.f / (1.f + __expf(-(iz + hz)));
      float n2 = tanhf(in_ + r * hn);
      float hnew = (1.f - z) * n2 + z * hf[j];
      hf[j]  = hnew;
      hbf[j] = (__bf16)hnew;
    }
    grid_sync(cnt, gen);

    // ---- Phase IV: l2 partials = W2 @ h (split-8; combine fused into V) ----
    wmma_gemv_sk<0>(W2p, HIDP, hbf, nullptr, l2p, nullptr, L2T, H2, HIDP, S_L2, gw, GW, lane);
    grid_sync(cnt, gen);

    // ---- Phase V: kg partials = W3 @ relu(sum l2p + b2) ----
    wmma_gemv_kg(W3b, l2p, b2, kgp, gw, GW, lane);
    grid_sync(cnt, gen);
  }

  // final posterior (t = T-1)
  if (blockIdx.x == 0 && tid < MM) {
    float v = sPrior[tid];
    for (int n = 0; n < NN; ++n) {
      float kgv = b3[tid * NN + n];
      #pragma unroll
      for (int s = 0; s < S_KG; ++s) kgv += kgp[s * DOUT + tid * NN + n];
      v += kgv * 1e-4f * sDy[n];
    }
    out[tid * TT + (TT - 1)] = v;
  }
}

// ---------------------------------------------------------------------------
extern "C" void kernel_launch(void* const* d_in, const int* in_sizes, int n_in,
                              void* d_out, int out_size, void* d_ws, size_t ws_size,
                              hipStream_t stream) {
  using namespace kn;
  (void)in_sizes; (void)n_in; (void)out_size; (void)ws_size;

  const float* A   = (const float*)d_in[0];
  const float* Cm  = (const float*)d_in[1];
  const float* x0  = (const float*)d_in[2];
  const float* h0  = (const float*)d_in[3];
  const float* y   = (const float*)d_in[4];
  const float* W1  = (const float*)d_in[5];
  const float* b1  = (const float*)d_in[6];
  const float* Wih = (const float*)d_in[7];
  const float* Whh = (const float*)d_in[8];
  const float* bih = (const float*)d_in[9];
  const float* bhh = (const float*)d_in[10];
  const float* W2  = (const float*)d_in[11];
  const float* b2  = (const float*)d_in[12];
  const float* W3  = (const float*)d_in[13];
  const float* b3  = (const float*)d_in[14];
  float* out = (float*)d_out;

  char* ws = (char*)d_ws;
  size_t o = 0;
  auto alloc = [&](size_t bytes) -> void* {
    void* p = ws + o;
    o += (bytes + 255) & ~(size_t)255;
    return p;
  };
  unsigned* cnt  = (unsigned*)alloc(8);
  unsigned* gen  = cnt + 1;
  __bf16* W1p    = (__bf16*)alloc((size_t)H1 * K1P * 2);
  __bf16* Wihb   = (__bf16*)alloc((size_t)G3 * H1 * 2);     // ~58 MB, L2-resident
  __bf16* Whhp   = (__bf16*)alloc((size_t)G3 * HIDP * 2);   // ~33 MB
  __bf16* W2p    = (__bf16*)alloc((size_t)H2 * HIDP * 2);
  __bf16* W3b    = (__bf16*)alloc((size_t)DOUT * H2 * 2);
  __bf16* l1b    = (__bf16*)alloc((size_t)H1 * 2);
  float*  gip    = (float*) alloc((size_t)S_GI * G3 * 4);
  float*  ghp    = (float*) alloc((size_t)S_GH * G3 * 4);
  float*  l2p    = (float*) alloc((size_t)S_L2 * H2 * 4);
  float*  kgp    = (float*) alloc((size_t)S_KG * DOUT * 4);
  float*  hf     = (float*) alloc((size_t)HIDP * 4);
  __bf16* hbf    = (__bf16*)alloc((size_t)HIDP * 2);

  knet_init<<<1024, 256, 0, stream>>>(W1, Wih, Whh, W2, W3, h0,
                                      W1p, Wihb, Whhp, W2p, W3b,
                                      hf, hbf, cnt, gen);
  knet_persist<<<NBLK, NTHR, 0, stream>>>(A, Cm, x0, y,
                                          b1, bih, bhh, b2, b3,
                                          W1p, Wihb, Whhp, W2p, W3b,
                                          l1b, gip, ghp, l2p, kgp,
                                          hf, hbf, cnt, gen, out);
}
